// CTKD_50491635532083
// MI455X (gfx1250) — matrix operand order repaired and checked by reading, not compiled, not run on hardware
//
#include <hip/hip_runtime.h>
#include <hip/hip_bf16.h>
#include <math.h>
#include <stdint.h>

typedef _Float16 v16h __attribute__((ext_vector_type(16)));
typedef _Float16 h8   __attribute__((ext_vector_type(8)));
typedef float    v8f  __attribute__((ext_vector_type(8)));
typedef float    f4   __attribute__((ext_vector_type(4)));

#define NUM_CLASSES 1000
#define FEAT        256
#define C_PAD       1008          // 63 tiles of 16 (padded rows are zero)
#define N_CTILES    63
#define INV_T       (1.0f / 0.07f)
#define EPSN        1e-12f

// LDS proto tile: 16 rows x 256 halfs, padded to 264 halfs (132 dwords) per
// row so the 16 lanes of a half-wave hit distinct banks (132 % 64 = 4).
#define LDS_PITCH_H 264

// ---- workspace layout (bytes, all 16B aligned) ----
#define OFF_SUMS_SRC  0
#define OFF_CNT_SRC   (OFF_SUMS_SRC + NUM_CLASSES * FEAT * 4)
#define OFF_SUMS_TGT  (OFF_CNT_SRC  + NUM_CLASSES * 4)
#define OFF_CNT_TGT   (OFF_SUMS_TGT + NUM_CLASSES * FEAT * 4)
#define OFF_PH_SRC    (OFF_CNT_TGT  + NUM_CLASSES * 4)
#define OFF_PH_TGT    (OFF_PH_SRC   + C_PAD * FEAT * 2)
#define OFF_ACCUM     (OFF_PH_TGT   + C_PAD * FEAT * 2)
#define WS_BYTES      (OFF_ACCUM + 16)

// ---------------------------------------------------------------------------
// Fragment loaders for V_WMMA_F32_16X16X32_F16 (per ISA 16-bit layouts).
// A (16x32, MxK): lane<16  holds K = base+0..7  and base+16..23 of row M=lane
//                 lane>=16 holds K = base+8..15 and base+24..31 of row M=lane-16
// B (32x16, KxN): lane<16  holds K = base+0..7  and base+16..23 of col N=lane
//                 lane>=16 holds K = base+8..15 and base+24..31 of col N=lane-16
// ---------------------------------------------------------------------------
__device__ inline v16h load_a_frag_h(const _Float16* __restrict__ row, int kk, int hi) {
  const int b = kk * 32;
  h8 lo = *(const h8*)(row + b + (hi ? 8 : 0));
  h8 hh = *(const h8*)(row + b + 16 + (hi ? 8 : 0));
  v16h r;
#pragma unroll
  for (int e = 0; e < 8; ++e) { r[e] = lo[e]; r[e + 8] = hh[e]; }
  return r;
}

__device__ inline v16h load_b_frag_h(const _Float16* __restrict__ row, int kk, int hi) {
  const int b = kk * 32 + (hi ? 8 : 0);
  h8 lo = *(const h8*)(row + b);
  h8 hh = *(const h8*)(row + b + 16);
  v16h r;
#pragma unroll
  for (int e = 0; e < 8; ++e) { r[e] = lo[e]; r[e + 8] = hh[e]; }
  return r;
}

// ---------------------------------------------------------------------------
// 1) Prototype segment-sum: atomic f32 adds into a 1 MB L2-resident table.
// ---------------------------------------------------------------------------
__global__ void proto_accum_kernel(const float* __restrict__ feats,
                                   const int* __restrict__ labels,
                                   float* __restrict__ sums,
                                   float* __restrict__ counts, int N) {
  int tid = blockIdx.x * blockDim.x + threadIdx.x;
  if (tid >= N * (FEAT / 4)) return;
  int row = tid >> 6;
  int ch  = tid & 63;
  int lab = labels[row];
  f4 v = ((const f4*)feats)[row * 64 + ch];
  float* dst = sums + lab * FEAT + ch * 4;
  atomicAdd(dst + 0, v[0]);
  atomicAdd(dst + 1, v[1]);
  atomicAdd(dst + 2, v[2]);
  atomicAdd(dst + 3, v[3]);
  if (ch == 0) atomicAdd(counts + lab, 1.0f);
}

// ---------------------------------------------------------------------------
// 2) Finalize prototypes; write f32 protos to d_out (src only) and the
//    L2-normalized f16 padded table to ws.
// ---------------------------------------------------------------------------
__global__ void proto_finalize_kernel(const float* __restrict__ sums,
                                      const float* __restrict__ counts,
                                      float* __restrict__ out_protos,  // may be null
                                      _Float16* __restrict__ protos_h) {
  int c = blockIdx.x;
  int t = threadIdx.x;  // 0..255
  float cnt = counts[c];
  float p = sums[c * FEAT + t] / fmaxf(cnt, 1.0f);
  if (out_protos) out_protos[c * FEAT + t] = p;

  float ss = p * p;
#pragma unroll
  for (int m = 16; m >= 1; m >>= 1) ss += __shfl_xor(ss, m, 32);
  __shared__ float red[8];
  if ((t & 31) == 0) red[t >> 5] = ss;
  __syncthreads();
  float tot = red[0] + red[1] + red[2] + red[3] + red[4] + red[5] + red[6] + red[7];

  float inv = 1.0f / fmaxf(sqrtf(tot), EPSN);
  protos_h[c * FEAT + t] = (_Float16)(p * inv);
}

// ---------------------------------------------------------------------------
// 3) Structure matrices via WMMA (one wave per 16x16 tile, src+tgt).
// ---------------------------------------------------------------------------
__global__ void struct_kernel(const _Float16* __restrict__ ph_src,
                              const _Float16* __restrict__ ph_tgt,
                              float* __restrict__ out_struct,
                              float* __restrict__ struct_accum) {
  int lane = threadIdx.x & 31;
  int hi = lane >> 4;
  int ln = lane & 15;
  int tm = blockIdx.x, tn = blockIdx.y;

  const _Float16* arow_s = ph_src + (size_t)(tm * 16 + ln) * FEAT;
  const _Float16* brow_s = ph_src + (size_t)(tn * 16 + ln) * FEAT;
  const _Float16* arow_t = ph_tgt + (size_t)(tm * 16 + ln) * FEAT;
  const _Float16* brow_t = ph_tgt + (size_t)(tn * 16 + ln) * FEAT;

  v8f accS = {};
  v8f accT = {};
#pragma unroll
  for (int kk = 0; kk < 8; ++kk) {
    v16h as = load_a_frag_h(arow_s, kk, hi);
    v16h bs = load_b_frag_h(brow_s, kk, hi);
    accS = __builtin_amdgcn_wmma_f32_16x16x32_f16(false, as, false, bs,
                                                  (short)0, accS, false, false);
    v16h at = load_a_frag_h(arow_t, kk, hi);
    v16h bt = load_b_frag_h(brow_t, kk, hi);
    accT = __builtin_amdgcn_wmma_f32_16x16x32_f16(false, at, false, bt,
                                                  (short)0, accT, false, false);
  }

  float local = 0.0f;
  int n = tn * 16 + ln;
#pragma unroll
  for (int i = 0; i < 8; ++i) {
    int m = tm * 16 + i + hi * 8;
    if (m < NUM_CLASSES && n < NUM_CLASSES) {
      float s = accS[i];
      out_struct[(size_t)m * NUM_CLASSES + n] = s;
      float d = s - accT[i];
      local += d * d;
    }
  }
#pragma unroll
  for (int m = 16; m >= 1; m >>= 1) local += __shfl_xor(local, m, 32);
  if (lane == 0) atomicAdd(struct_accum, local);
}

// ---------------------------------------------------------------------------
// Async DMA of one 16-byte chunk of a proto tile into LDS (ASYNCcnt-tracked).
// chunk c of tile ct: row = c>>5 (32 x 16B per 512B row), colByte = (c&31)*16.
// ---------------------------------------------------------------------------
__device__ inline void async_tile_chunk(const _Float16* __restrict__ protos_h,
                                        unsigned lds_base, int ct, int c) {
  int row = c >> 5;
  int colB = (c & 31) * 16;
  uint64_t ga = (uint64_t)(uintptr_t)((const char*)protos_h +
                                      ((size_t)(ct * 16 + row) * (FEAT * 2) + colB));
  unsigned la = lds_base + (unsigned)(row * (LDS_PITCH_H * 2) + colB);
  asm volatile("global_load_async_to_lds_b128 %0, %1, off"
               :: "v"(la), "v"(ga) : "memory");
}

// ---------------------------------------------------------------------------
// 4) Contrastive loss via WMMA. One wave owns a 16-row feature block; the 8
//    waves of a block share each 16-column proto tile, double-buffered in LDS
//    via global_load_async_to_lds_b128 (issue ct+1, compute ct, wait+barrier).
//    Row normalization folded into the logits; no-max softmax is safe since
//    |logit| <= 1/T. Inactive tail waves clamp rows and keep EXEC all-ones.
// ---------------------------------------------------------------------------
__global__ void __launch_bounds__(256)
contrast_kernel(const float* __restrict__ feats,
                const int* __restrict__ labels,
                const _Float16* __restrict__ protos_h,
                float* __restrict__ contrast_accum, int N) {
  __shared__ _Float16 ldsbuf[2][16][LDS_PITCH_H];

  int t = threadIdx.x;
  int lane = t & 31;
  int wave = t >> 5;
  int numRB = N >> 4;                       // N = 200000 -> 12500 exact
  int rb = blockIdx.x * 8 + wave;           // wave-uniform
  bool active = (rb < numRB);
  int rbc = active ? rb : (numRB - 1);      // clamp; inactive waves recompute

  int hi = lane >> 4;
  int ln = lane & 15;
  int r0 = rbc * 16;

  // ---- A fragments: raw f32 feats -> f16, plus per-lane sum of squares ----
  const float* frow = feats + (size_t)(r0 + ln) * FEAT;
  v16h a[8];
  float ss = 0.0f;
#pragma unroll
  for (int kk = 0; kk < 8; ++kk) {
    int b = kk * 32;
    f4 x0 = *(const f4*)(frow + b + hi * 8);
    f4 x1 = *(const f4*)(frow + b + hi * 8 + 4);
    f4 x2 = *(const f4*)(frow + b + 16 + hi * 8);
    f4 x3 = *(const f4*)(frow + b + 16 + hi * 8 + 4);
    v16h r;
#pragma unroll
    for (int e = 0; e < 4; ++e) {
      r[e]      = (_Float16)x0[e];
      r[e + 4]  = (_Float16)x1[e];
      r[e + 8]  = (_Float16)x2[e];
      r[e + 12] = (_Float16)x3[e];
      ss += x0[e] * x0[e] + x1[e] * x1[e] + x2[e] * x2[e] + x3[e] * x3[e];
    }
    a[kk] = r;
  }
  ss += __shfl_xor(ss, 16, 32);             // lanes L and L+16 cover row L&15
  float inv = 1.0f / fmaxf(sqrtf(ss), EPSN);

  float invm[8];
  int lab[8];
#pragma unroll
  for (int i = 0; i < 8; ++i) {
    int m = i + hi * 8;
    invm[i] = __shfl(inv, m, 32);           // row m's inv lives in lane m
    lab[i] = labels[r0 + m];
  }

  float sumexp[8], tgt[8];
#pragma unroll
  for (int i = 0; i < 8; ++i) { sumexp[i] = 0.0f; tgt[i] = 0.0f; }

  unsigned lbase[2];
  lbase[0] = (unsigned)(uintptr_t)&ldsbuf[0][0][0];   // low 32 bits = LDS offset
  lbase[1] = (unsigned)(uintptr_t)&ldsbuf[1][0][0];

  // Prime buffer 0 with tile 0 (two 16B chunks per thread, 512 total).
  async_tile_chunk(protos_h, lbase[0], 0, t);
  async_tile_chunk(protos_h, lbase[0], 0, t + 256);
  asm volatile("s_wait_asynccnt 0x0" ::: "memory");
  __syncthreads();

  for (int ct = 0; ct < N_CTILES; ++ct) {
    // Kick off the async DMA for the next tile into the other buffer; the
    // TDM-style copy overlaps with the WMMA sequence below.
    if (ct + 1 < N_CTILES) {
      unsigned nb = lbase[(ct + 1) & 1];
      async_tile_chunk(protos_h, nb, ct + 1, t);
      async_tile_chunk(protos_h, nb, ct + 1, t + 256);
    }

    const _Float16* prow = &ldsbuf[ct & 1][ln][0];
    v8f acc = {};
#pragma unroll
    for (int kk = 0; kk < 8; ++kk) {
      v16h bfrag = load_b_frag_h(prow, kk, hi);
      acc = __builtin_amdgcn_wmma_f32_16x16x32_f16(false, a[kk], false, bfrag,
                                                   (short)0, acc, false, false);
    }

    int col = ct * 16 + ln;
    bool valid = (col < NUM_CLASSES);       // mask padded columns 1000..1007
#pragma unroll
    for (int i = 0; i < 8; ++i) {
      float logit = acc[i] * invm[i] * INV_T;
      if (valid) {
        sumexp[i] += __expf(logit);
        if (col == lab[i]) tgt[i] += logit; // exactly one hit per row
      }
    }

    // Our async loads done -> barrier makes everyone's tile visible, and
    // guarantees all waves finished reading buf[ct&1] before it is reused.
    asm volatile("s_wait_asynccnt 0x0" ::: "memory");
    __syncthreads();
  }

  // Reduce across the 16 lanes of each half-wave (rows differ across halves).
#pragma unroll
  for (int i = 0; i < 8; ++i) {
#pragma unroll
    for (int m = 8; m >= 1; m >>= 1) {
      sumexp[i] += __shfl_xor(sumexp[i], m, 32);
      tgt[i]    += __shfl_xor(tgt[i], m, 32);
    }
  }
  if (ln == 0 && active) {
    float part = 0.0f;
#pragma unroll
    for (int i = 0; i < 8; ++i) part += logf(sumexp[i]) - tgt[i];
    atomicAdd(contrast_accum, part);        // lanes 0 & 16: rows 0..7 / 8..15
  }
}

// ---------------------------------------------------------------------------
// 5) Finalize the two scalar losses.
// ---------------------------------------------------------------------------
__global__ void finalize_kernel(const float* __restrict__ accum,
                                float* __restrict__ out, float invCC, float invN) {
  out[0] = accum[0] * invCC;   // struct_loss = mean over C*C
  out[1] = accum[1] * invN;    // contrast_loss = mean over N
}

extern "C" void kernel_launch(void* const* d_in, const int* in_sizes, int n_in,
                              void* d_out, int out_size, void* d_ws, size_t ws_size,
                              hipStream_t stream) {
  const float* src_feats  = (const float*)d_in[0];
  const int*   src_labels = (const int*)d_in[1];
  const float* tgt_feats  = (const float*)d_in[2];
  const int*   tgt_labels = (const int*)d_in[3];
  const int N = in_sizes[0] / FEAT;

  char* ws = (char*)d_ws;
  float*    sums_s = (float*)(ws + OFF_SUMS_SRC);
  float*    cnt_s  = (float*)(ws + OFF_CNT_SRC);
  float*    sums_t = (float*)(ws + OFF_SUMS_TGT);
  float*    cnt_t  = (float*)(ws + OFF_CNT_TGT);
  _Float16* ph_s   = (_Float16*)(ws + OFF_PH_SRC);
  _Float16* ph_t   = (_Float16*)(ws + OFF_PH_TGT);
  float*    accum  = (float*)(ws + OFF_ACCUM);     // [0]=struct, [1]=contrast

  float* out        = (float*)d_out;
  float* out_protos = out + 2;                       // [1000,256]
  float* out_struct = out + 2 + NUM_CLASSES * FEAT;  // [1000,1000]

  // Zero accumulators + padded proto tables (re-runs every graph replay).
  hipMemsetAsync(d_ws, 0, WS_BYTES, stream);

  int accThreads = N * (FEAT / 4);
  int accBlocks = (accThreads + 255) / 256;
  proto_accum_kernel<<<accBlocks, 256, 0, stream>>>(src_feats, src_labels, sums_s, cnt_s, N);
  proto_accum_kernel<<<accBlocks, 256, 0, stream>>>(tgt_feats, tgt_labels, sums_t, cnt_t, N);

  proto_finalize_kernel<<<NUM_CLASSES, FEAT, 0, stream>>>(sums_s, cnt_s, out_protos, ph_s);
  proto_finalize_kernel<<<NUM_CLASSES, FEAT, 0, stream>>>(sums_t, cnt_t, nullptr, ph_t);

  struct_kernel<<<dim3(N_CTILES, N_CTILES), 32, 0, stream>>>(ph_s, ph_t, out_struct, accum + 0);

  int numRB = N >> 4;                         // 12500 waves, 8 waves per block
  contrast_kernel<<<(numRB + 7) / 8, 256, 0, stream>>>(src_feats, src_labels, ph_s, accum + 1, N);

  finalize_kernel<<<1, 1, 0, stream>>>(accum, out,
                                       1.0f / ((float)NUM_CLASSES * (float)NUM_CLASSES),
                                       1.0f / (float)N);
}
